// SDCNModel_66039417143760
// MI455X (gfx1250) — matrix operand order repaired and checked by reading, not compiled, not run on hardware
//
#include <hip/hip_runtime.h>

// ---------------------------------------------------------------------------
// Types for CDNA5 WMMA (gfx1250, wave32)
// ---------------------------------------------------------------------------
typedef __attribute__((ext_vector_type(16))) __bf16         v16bf;
typedef __attribute__((ext_vector_type(16))) unsigned short v16u;
typedef __attribute__((ext_vector_type(8)))  float          v8f;
typedef __attribute__((ext_vector_type(4)))  unsigned int   u32x4;
typedef __attribute__((ext_vector_type(4)))  int            i32x4;
typedef __attribute__((ext_vector_type(8)))  int            i32x8;

#if defined(__has_builtin)
#  if __has_builtin(__builtin_amdgcn_tensor_load_to_lds)
#    define HAVE_TDM 1
#  else
#    define HAVE_TDM 0
#  endif
#else
#  define HAVE_TDM 0
#endif

__device__ __forceinline__ unsigned short f32_to_bf16(float f) {
  unsigned int u = __float_as_uint(f);
  u += 0x7fffu + ((u >> 16) & 1u);          // round-to-nearest-even
  return (unsigned short)(u >> 16);
}

#if HAVE_TDM
// Issue one TDM 2D tile load: tile_rows x 64 bf16 elements, row stride kstride,
// into LDS at lds_off with 8-half row padding (LDS row stride 72 halfs).
// Rows beyond rows_rem read as zero (TDM OOB semantics) -> replaces M guards.
__device__ __forceinline__ void tdm_load_tile(const unsigned short* gptr,
                                              unsigned lds_off, int kstride,
                                              int rows_rem, int tile_rows) {
  unsigned long long ga = (unsigned long long)(const void*)gptr;
  u32x4 g0;
  g0[0] = 1u;                                   // count=1, user descriptor
  g0[1] = lds_off;                              // lds_addr (bytes)
  g0[2] = (unsigned)ga;                         // global_addr[31:0]
  g0[3] = (unsigned)((ga >> 32) & 0x01ffffffull) | (2u << 30); // addr[56:32]|type=2
  i32x8 g1;
  // data_size=2B(1), pad_enable, pad_interval=32 DWORDs(4), pad_amount=4 DWORDs(3)
  g1[0] = (int)((1u << 16) | (1u << 20) | (4u << 22) | (3u << 25));
  g1[1] = (int)(((unsigned)kstride & 0xffffu) << 16);              // tensor_dim0 lo16
  g1[2] = (int)((((unsigned)kstride >> 16) & 0xffffu) |
                (((unsigned)rows_rem & 0xffffu) << 16));           // dim0 hi | dim1 lo
  g1[3] = (int)((((unsigned)rows_rem >> 16) & 0xffffu) | (64u << 16)); // dim1 hi | tile_dim0=64
  g1[4] = tile_rows;                            // tile_dim1, tile_dim2=0
  g1[5] = kstride;                              // tensor_dim0_stride lo32
  g1[6] = 0;
  g1[7] = 0;
  i32x4 z4 = {0, 0, 0, 0};
#if __clang_major__ >= 23
  i32x8 z8 = {0, 0, 0, 0, 0, 0, 0, 0};
  __builtin_amdgcn_tensor_load_to_lds(g0, g1, z4, z4, z8, 0);
#else
  __builtin_amdgcn_tensor_load_to_lds(g0, g1, z4, z4, 0);
#endif
}
#endif

// ---------------------------------------------------------------------------
// bf16 WMMA GEMM:  out = act( A[M,Kp] @ W[Kp,Np] + bias )
//   A  bf16 row-major [M, Kp]
//   Bt bf16 row-major [Np, Kp]  (pre-transposed weights -> contiguous K)
//   outb: bf16 [M, ldout] (padding cols zeroed)   (nullable)
//   outf: f32  [M, ldf], cols < Nreal             (nullable)
// Block tile 128x64, 8 waves; each wave: 32x32 region = 2x2 WMMA tiles.
// K staged 64 at a time; A and B tiles staged by the Tensor Data Mover,
// double-buffered with counted s_wait_tensorcnt for DMA/WMMA overlap.
// ---------------------------------------------------------------------------
#define BM 128
#define BN 64
#define KT 64
#define LSTR 72   // LDS row stride in halfs (64 + 8 pad => 144B rows, 16B aligned)

#if HAVE_TDM
#define NBUF 2
#else
#define NBUF 1
#endif

__global__ __launch_bounds__(256) void gemm_bf16_wmma(
    const unsigned short* __restrict__ A,
    const unsigned short* __restrict__ Bt,
    const float* __restrict__ bias,
    unsigned short* __restrict__ outb, int ldout,
    float* __restrict__ outf, int ldf,
    int M, int Kp, int Np, int Nreal, int relu)
{
  __shared__ __align__(16) unsigned short As[NBUF][BM * LSTR];
  __shared__ __align__(16) unsigned short Bs[NBUF][BN * LSTR];

  const int tid  = threadIdx.x;
  const int lane = tid & 31;
  const int wave = tid >> 5;
  const int wm   = wave & 3;   // 4 waves along M
  const int wn   = wave >> 2;  // 2 waves along N
  const int m0   = blockIdx.x * BM;
  const int n0   = blockIdx.y * BN;

  v8f acc[2][2] = {};

#if HAVE_TDM
  const unsigned ldsA[2] = { (unsigned)(unsigned long long)&As[0][0],
                             (unsigned)(unsigned long long)&As[NBUF - 1][0] };
  const unsigned ldsB[2] = { (unsigned)(unsigned long long)&Bs[0][0],
                             (unsigned)(unsigned long long)&Bs[NBUF - 1][0] };
  // prologue: stage first K slab into buffer 0
  if (tid < 32) {
    tdm_load_tile(A  + (long long)m0 * Kp, ldsA[0], Kp, M  - m0, BM);
    tdm_load_tile(Bt + (long long)n0 * Kp, ldsB[0], Kp, Np - n0, BN);
  }
#else
  const int srow = tid >> 3;        // 0..31
  const int sc8  = (tid & 7) << 3;  // 0..56
#endif

  int cur = 0;
  for (int kt = 0; kt < Kp; kt += KT) {
#if HAVE_TDM
    if (tid < 32) {
      int nkt = kt + KT;
      if (nkt < Kp) {
        // issue next slab into the other buffer, then wait only for the
        // two OLDEST tensor ops (the current slab): DMA overlaps WMMA.
        tdm_load_tile(A  + (long long)m0 * Kp + nkt, ldsA[cur ^ 1], Kp, M  - m0, BM);
        tdm_load_tile(Bt + (long long)n0 * Kp + nkt, ldsB[cur ^ 1], Kp, Np - n0, BN);
        __builtin_amdgcn_s_wait_tensorcnt(2);
      } else {
        __builtin_amdgcn_s_wait_tensorcnt(0);
      }
    }
    __syncthreads();   // current slab visible to all waves
#else
    __syncthreads();   // previous slab fully consumed before restaging
    #pragma unroll
    for (int rr = 0; rr < BM; rr += 32) {
      int row  = rr + srow;
      int grow = m0 + row;
      uint4 v = {0u, 0u, 0u, 0u};
      if (grow < M)
        v = *(const uint4*)(A + (long long)grow * Kp + kt + sc8);
      *(uint4*)(&As[0][row * LSTR + sc8]) = v;
    }
    #pragma unroll
    for (int rr = 0; rr < BN; rr += 32) {
      uint4 v = *(const uint4*)(Bt + (long long)(n0 + rr + srow) * Kp + kt + sc8);
      *(uint4*)(&Bs[0][(rr + srow) * LSTR + sc8]) = v;
    }
    __syncthreads();
#endif

    // ---- two WMMA k-substeps over the staged 64-wide K slab ----
    #pragma unroll
    for (int ks = 0; ks < 2; ++ks) {
      v16u au[2], bu[2];
      const int khalf = (lane >> 4) << 4;   // lanes 0-15: K 0-15 / 16-31: K 16-31
      #pragma unroll
      for (int tm = 0; tm < 2; ++tm) {
        const unsigned short* ap =
            &As[cur][(wm * 32 + tm * 16 + (lane & 15)) * LSTR + ks * 32 + khalf];
        #pragma unroll
        for (int i = 0; i < 16; ++i) au[tm][i] = ap[i];
      }
      const int koff = (lane >> 4) << 4;    // lanes 0-15: K 0-15 / 16-31: K 16-31
      #pragma unroll
      for (int tn = 0; tn < 2; ++tn) {
        const unsigned short* bp =
            &Bs[cur][(wn * 32 + tn * 16 + (lane & 15)) * LSTR + ks * 32 + koff];
        #pragma unroll
        for (int i = 0; i < 16; ++i) bu[tn][i] = bp[i];   // contiguous -> 2x ds_load_b128
      }
      #pragma unroll
      for (int tm = 0; tm < 2; ++tm)
        #pragma unroll
        for (int tn = 0; tn < 2; ++tn)
          acc[tm][tn] = __builtin_amdgcn_wmma_f32_16x16x32_bf16(
              false, __builtin_bit_cast(v16bf, au[tm]),
              false, __builtin_bit_cast(v16bf, bu[tn]),
              (short)0, acc[tm][tn], false, false);
    }

#if HAVE_TDM
    __syncthreads();   // all waves done reading buf[cur]; safe to overwrite next iter
    cur ^= 1;
#endif
  }

  // ---- epilogue: bias + relu, dual store ----
  const int mhalf = (lane >> 4) << 3;   // lane<16 -> rows 0-7, else rows 8-15
  #pragma unroll
  for (int tm = 0; tm < 2; ++tm) {
    #pragma unroll
    for (int tn = 0; tn < 2; ++tn) {
      int  col = n0 + wn * 32 + tn * 16 + (lane & 15);
      bool inN = col < Nreal;
      float bv = (bias != nullptr && inN) ? bias[col] : 0.0f;
      #pragma unroll
      for (int r = 0; r < 8; ++r) {
        int row = m0 + wm * 32 + tm * 16 + mhalf + r;
        if (row < M) {
          float v = acc[tm][tn][r] + bv;
          if (relu) v = v > 0.0f ? v : 0.0f;
          if (outb) outb[(long long)row * ldout + col] =
              inN ? f32_to_bf16(v) : (unsigned short)0;
          if (outf && inN) outf[(long long)row * ldf + col] = v;
        }
      }
    }
  }
}

// ---------------------------------------------------------------------------
// Helper kernels
// ---------------------------------------------------------------------------
__global__ void cvt_pad_bf16(const float* __restrict__ in,
                             unsigned short* __restrict__ out,
                             int R, int C, int Rp, int Cp) {
  long long i = (long long)blockIdx.x * blockDim.x + threadIdx.x;
  if (i >= (long long)Rp * Cp) return;
  int r = (int)(i / Cp), c = (int)(i % Cp);
  float v = (r < R && c < C) ? in[(long long)r * C + c] : 0.0f;
  out[i] = f32_to_bf16(v);
}

// transpose+pad weights: in [R=din, C=dout] f32 -> out [Cp, Rp] bf16 (out[n][k]=in[k][n])
__global__ void cvt_pad_bf16_T(const float* __restrict__ in,
                               unsigned short* __restrict__ out,
                               int R, int C, int Rp, int Cp) {
  long long i = (long long)blockIdx.x * blockDim.x + threadIdx.x;
  if (i >= (long long)Cp * Rp) return;
  int n = (int)(i / Rp), k = (int)(i % Rp);
  float v = (k < R && n < C) ? in[(long long)k * C + n] : 0.0f;
  out[i] = f32_to_bf16(v);
}

__global__ void fill_f32(float* __restrict__ p, long long n, float v) {
  long long i = (long long)blockIdx.x * blockDim.x + threadIdx.x;
  if (i < n) p[i] = v;
}

__global__ void deg_kernel(const int* __restrict__ ei, float* __restrict__ deg, int E) {
  int e = blockIdx.x * blockDim.x + threadIdx.x;
  if (e < E) atomicAdd(&deg[ei[E + e]], 1.0f);
}

__global__ void dinv_kernel(float* __restrict__ d, int n) {
  int i = blockIdx.x * blockDim.x + threadIdx.x;
  if (i < n) d[i] = rsqrtf(d[i] + 1.0f);
}

__global__ void scatter_kernel(const int* __restrict__ ei,
                               const float* __restrict__ dinv,
                               const float* __restrict__ h,
                               float* __restrict__ agg, int E) {
  long long t = (long long)blockIdx.x * blockDim.x + threadIdx.x;
  if (t >= (long long)E * 128) return;
  int e = (int)(t >> 7), f = (int)(t & 127);
  int s = ei[e], d = ei[E + e];
  float w = dinv[s] * dinv[d];
  atomicAdd(&agg[(long long)d * 128 + f], h[(long long)s * 128 + f] * w);
}

// gcn_in = 0.5*ae + 0.5*(agg + h*dinv^2 + b)  -> bf16
__global__ void combine1_kernel(const float* __restrict__ agg,
                                const float* __restrict__ h,
                                const float* __restrict__ dinv,
                                const float* __restrict__ b,
                                const float* __restrict__ ae,
                                unsigned short* __restrict__ gcb, int n) {
  long long t = (long long)blockIdx.x * blockDim.x + threadIdx.x;
  if (t >= (long long)n * 128) return;
  int node = (int)(t >> 7), f = (int)(t & 127);
  float di = dinv[node];
  float conv = agg[t] + h[t] * di * di + b[f];
  gcb[t] = f32_to_bf16(0.5f * ae[t] + 0.5f * conv);
}

// z = agg + h*dinv^2 + b  -> f32
__global__ void combine2_kernel(const float* __restrict__ agg,
                                const float* __restrict__ h,
                                const float* __restrict__ dinv,
                                const float* __restrict__ b,
                                float* __restrict__ z, int n) {
  long long t = (long long)blockIdx.x * blockDim.x + threadIdx.x;
  if (t >= (long long)n * 128) return;
  int node = (int)(t >> 7), f = (int)(t & 127);
  float di = dinv[node];
  z[t] = agg[t] + h[t] * di * di + b[f];
}

// one wave32 per node: student-t soft assignment over C=10, D=128
__global__ __launch_bounds__(256) void q_kernel(const float* __restrict__ z,
                                                const float* __restrict__ centers,
                                                float* __restrict__ q, int n) {
  int node = blockIdx.x * 8 + (threadIdx.x >> 5);
  int lane = threadIdx.x & 31;
  if (node >= n) return;
  const float* zr = z + (long long)node * 128;
  float zv[4];
  #pragma unroll
  for (int j = 0; j < 4; ++j) zv[j] = zr[lane * 4 + j];
  float qc[10];
  float s = 0.0f;
  #pragma unroll
  for (int c = 0; c < 10; ++c) {
    const float* ce = centers + c * 128;
    float p = 0.0f;
    #pragma unroll
    for (int j = 0; j < 4; ++j) {
      float d = zv[j] - ce[lane * 4 + j];
      p += d * d;
    }
    #pragma unroll
    for (int off = 16; off > 0; off >>= 1) p += __shfl_xor(p, off, 32);
    qc[c] = 1.0f / (1.0f + p);
    s += qc[c];
  }
  if (lane == 0) {
    float inv = 1.0f / s;
    #pragma unroll
    for (int c = 0; c < 10; ++c) q[(long long)node * 10 + c] = qc[c] * inv;
  }
}

// ---------------------------------------------------------------------------
// Orchestration
// ---------------------------------------------------------------------------
extern "C" void kernel_launch(void* const* d_in, const int* in_sizes, int n_in,
                              void* d_out, int out_size, void* d_ws, size_t ws_size,
                              hipStream_t stream) {
  (void)n_in; (void)out_size; (void)ws_size;

  const float* x  = (const float*)d_in[0];
  const int*   ei = (const int*)d_in[1];
  const float* W[8]; const float* bvec[8];
  for (int i = 0; i < 8; ++i) {
    W[i]    = (const float*)d_in[2 + 2 * i];
    bvec[i] = (const float*)d_in[3 + 2 * i];
  }
  const float* Wg1 = (const float*)d_in[18];
  const float* bg1 = (const float*)d_in[19];
  const float* Wg2 = (const float*)d_in[20];
  const float* bg2 = (const float*)d_in[21];
  const float* centers = (const float*)d_in[22];

  const int Nn = in_sizes[0] / 1000;   // 50000
  const int E  = in_sizes[1] / 2;      // 800000

  float* out_ae = (float*)d_out;
  float* out_xh = out_ae + (long long)Nn * 128;
  float* out_z  = out_xh + (long long)Nn * 1000;
  float* out_q  = out_z  + (long long)Nn * 128;

  // workspace carve-outs (with reuse)
  char* p = (char*)d_ws;
  auto alloc = [&](size_t bytes) {
    void* r = (void*)p;
    p += (bytes + 255) & ~(size_t)255;
    return r;
  };
  unsigned short* Xb  = (unsigned short*)alloc((size_t)Nn * 1024 * 2);
  unsigned short* h1b = (unsigned short*)alloc((size_t)Nn * 512  * 2);
  unsigned short* h2b = (unsigned short*)alloc((size_t)Nn * 512  * 2);
  unsigned short* h3b = (unsigned short*)alloc((size_t)Nn * 2048 * 2);
  unsigned short* aeb = (unsigned short*)alloc((size_t)Nn * 128  * 2);
  unsigned short* gcb = (unsigned short*)alloc((size_t)Nn * 128  * 2);
  float* hg   = (float*)alloc((size_t)Nn * 128 * 4);
  float* agg  = (float*)alloc((size_t)Nn * 128 * 4);
  float* dinv = (float*)alloc((size_t)Nn * 4);

  const int wr[10] = {1000, 500, 500, 2000, 128, 2000, 500, 500, 128, 128};
  const int wc[10] = {500, 500, 2000, 128, 2000, 500, 500, 1000, 128, 128};
  const float* Wsrc[10] = {W[0], W[1], W[2], W[3], W[4], W[5], W[6], W[7], Wg1, Wg2};
  unsigned short* Wb[10];
  int wrp[10], wcp[10];
  for (int i = 0; i < 10; ++i) {
    wrp[i] = (wr[i] + 63) & ~63;
    wcp[i] = (wc[i] + 63) & ~63;
    Wb[i]  = (unsigned short*)alloc((size_t)wrp[i] * wcp[i] * 2);  // stored [wcp, wrp]
  }

  // ---- f32 -> bf16 conversions (zero padded; weights transposed) ----
  {
    long long tot = (long long)Nn * 1024;
    cvt_pad_bf16<<<(int)((tot + 255) / 256), 256, 0, stream>>>(x, Xb, Nn, 1000, Nn, 1024);
  }
  for (int i = 0; i < 10; ++i) {
    long long tot = (long long)wrp[i] * wcp[i];
    cvt_pad_bf16_T<<<(int)((tot + 255) / 256), 256, 0, stream>>>(
        Wsrc[i], Wb[i], wr[i], wc[i], wrp[i], wcp[i]);
  }

  const int gM = (Nn + BM - 1) / BM;
  auto gemm = [&](const unsigned short* A, int wi, const float* bias,
                  unsigned short* outb, float* outf, int ldf, int Nreal, int relu) {
    dim3 g(gM, wcp[wi] / BN);
    gemm_bf16_wmma<<<g, 256, 0, stream>>>(A, Wb[wi], bias, outb, wcp[wi],
                                          outf, ldf, Nn, wrp[wi], wcp[wi], Nreal, relu);
  };

  // ---- encoder ----
  gemm(Xb,  0, bvec[0], h1b, nullptr, 0,   500,  1);
  gemm(h1b, 1, bvec[1], h2b, nullptr, 0,   500,  1);
  gemm(h2b, 2, bvec[2], h3b, nullptr, 0,   2000, 1);
  gemm(h3b, 3, bvec[3], aeb, out_ae,  128, 128,  1);            // ae_embed
  // ---- decoder (reuse activation buffers) ----
  gemm(aeb, 4, bvec[4], h3b, nullptr, 0,   2000, 1);
  gemm(h3b, 5, bvec[5], h1b, nullptr, 0,   500,  1);
  gemm(h1b, 6, bvec[6], h2b, nullptr, 0,   500,  1);
  gemm(h2b, 7, bvec[7], nullptr, out_xh, 1000, 1000, 1);        // x_hat

  // ---- GCN conv 1: h = ae @ Wg1 ----
  gemm(aeb, 8, nullptr, nullptr, hg, 128, 128, 0);
  fill_f32<<<(Nn + 255) / 256, 256, 0, stream>>>(dinv, (long long)Nn, 0.0f);
  deg_kernel<<<(E + 255) / 256, 256, 0, stream>>>(ei, dinv, E);
  dinv_kernel<<<(Nn + 255) / 256, 256, 0, stream>>>(dinv, Nn);
  {
    long long tot = (long long)Nn * 128;
    fill_f32<<<(int)((tot + 255) / 256), 256, 0, stream>>>(agg, tot, 0.0f);
  }
  {
    long long tot = (long long)E * 128;
    scatter_kernel<<<(int)((tot + 255) / 256), 256, 0, stream>>>(ei, dinv, hg, agg, E);
  }
  {
    long long tot = (long long)Nn * 128;
    combine1_kernel<<<(int)((tot + 255) / 256), 256, 0, stream>>>(
        agg, hg, dinv, bg1, out_ae, gcb, Nn);
  }

  // ---- GCN conv 2: h = gcn_in @ Wg2 ----
  gemm(gcb, 9, nullptr, nullptr, hg, 128, 128, 0);
  {
    long long tot = (long long)Nn * 128;
    fill_f32<<<(int)((tot + 255) / 256), 256, 0, stream>>>(agg, tot, 0.0f);
  }
  {
    long long tot = (long long)E * 128;
    scatter_kernel<<<(int)((tot + 255) / 256), 256, 0, stream>>>(ei, dinv, hg, agg, E);
  }
  {
    long long tot = (long long)Nn * 128;
    combine2_kernel<<<(int)((tot + 255) / 256), 256, 0, stream>>>(
        agg, hg, dinv, bg2, out_z, Nn);
  }

  // ---- soft assignment q ----
  q_kernel<<<(Nn + 7) / 8, 256, 0, stream>>>(out_z, centers, out_q, Nn);
}
